// CasualAttention_13460427505741
// MI455X (gfx1250) — matrix operand order repaired and checked by reading, not compiled, not run on hardware
//
#include <hip/hip_runtime.h>
#include <hip/hip_bf16.h>

// ---------------------------------------------------------------------------
// Causal attention for MI455X (gfx1250, wave32, WMMA + async-to-LDS).
//   x   [4,2048,1024] f32, W_{q,k,v} [1024,1024] f32 -> out [4,2048,1024] f32
// Compute-bound (~120 GFLOP vs ~6us of HBM traffic at 23.3 TB/s) -> bf16
// WMMA path, fp32 accumulation, fp32 online softmax.
// Workspace (48 MB): Q bf16 [8192,1024], K bf16 [8192,1024],
//                    Vt bf16 [4,1024,2048] (transposed V for contiguous
//                    B-fragment loads in the P@V WMMA).
// ---------------------------------------------------------------------------

typedef __attribute__((ext_vector_type(16))) __bf16    v16bf;
typedef __attribute__((ext_vector_type(8)))  float     v8f;
typedef __attribute__((ext_vector_type(4)))  unsigned  u32x4;

union FragBF {
    v16bf          v;
    __bf16         h[16];
    unsigned short u[16];
    u32x4          q[2];
};

// Native f32 -> bf16 (compiler emits hardware cvt on gfx1250)
__device__ __forceinline__ unsigned short f2bf_bits(float f) {
    __bf16 h = (__bf16)f;
    return __builtin_bit_cast(unsigned short, h);
}

__device__ __forceinline__ v8f wmma_bf16(v16bf a, v16bf b, v8f c) {
    // D = A(16x32 bf16) x B(32x16 bf16) + C(16x16 f32)
    return __builtin_amdgcn_wmma_f32_16x16x32_bf16(
        /*neg_a=*/false, a, /*neg_b=*/false, b,
        /*c_mod=*/(short)0, c, /*reuse_a=*/false, /*reuse_b=*/false);
}

// ---------------------------------------------------------------------------
// Kernel 1: Y = x @ W^T for W in {Wq,Wk,Wv}; one wave = one 16x16 tile.
// A-frag (16x32): lane m=lane&15, kgrp=lane>>4;
//   elems 0..7  -> K = kk + kgrp*8 + e ; elems 8..15 -> K = kk+16+kgrp*8+(e-8)
// B-frag (32x16): lane n=lane&15, same K mapping as the A-frag
//   (B[k][n] = W[n][k] -> two runs of 8 floats of row n of W, 16 apart).
// ---------------------------------------------------------------------------
__global__ __launch_bounds__(128) void qkv_gemm_kernel(
    const float* __restrict__ x,
    const float* __restrict__ Wq,
    const float* __restrict__ Wk,
    const float* __restrict__ Wv,
    unsigned short* __restrict__ Qws,
    unsigned short* __restrict__ Kws,
    unsigned short* __restrict__ Vtws)
{
    const int tid   = threadIdx.x;
    const int lane  = tid & 31;
    const int wave  = tid >> 5;
    const int lhalf = lane & 15;
    const int kgrp  = lane >> 4;

    const int m0  = blockIdx.x * 16;                 // row tile in [0,8192)
    const int n0  = (blockIdx.y * 4 + wave) * 16;    // col tile in [0,1024)
    const int mat = blockIdx.z;                      // 0=Q 1=K 2=V

    const float* W    = (mat == 0) ? Wq : ((mat == 1) ? Wk : Wv);
    const float* arow = x + (size_t)(m0 + lhalf) * 1024;
    const float* brow = W + (size_t)(n0 + lhalf) * 1024;

    v8f acc = {};
    for (int kk = 0; kk < 1024; kk += 32) {
        FragBF a, b;
        const float* ap0 = arow + kk + kgrp * 8;
        const float* ap1 = arow + kk + 16 + kgrp * 8;
#pragma unroll
        for (int e = 0; e < 8; ++e) {
            a.h[e]     = (__bf16)ap0[e];
            a.h[e + 8] = (__bf16)ap1[e];
        }
        const float* bp = brow + kk + kgrp * 8;
#pragma unroll
        for (int e = 0; e < 8; ++e) { b.h[e] = (__bf16)bp[e]; b.h[e + 8] = (__bf16)bp[e + 16]; }

        if (kk + 32 < 1024) {            // -> global_prefetch_b8
            __builtin_prefetch(ap0 + 32, 0, 1);
            __builtin_prefetch(bp + 32, 0, 1);
        }
        acc = wmma_bf16(a.v, b.v, acc);
    }

    // C/D layout: element v -> M = v + 8*kgrp, N = n0 + lhalf
    if (mat < 2) {
        unsigned short* dst = (mat == 0) ? Qws : Kws;
#pragma unroll
        for (int v = 0; v < 8; ++v) {
            const int M = v + 8 * kgrp;
            dst[(size_t)(m0 + M) * 1024 + n0 + lhalf] = f2bf_bits(acc[v]);
        }
    } else {
        // V stored transposed: Vt[b][d][seq]
#pragma unroll
        for (int v = 0; v < 8; ++v) {
            const int M   = v + 8 * kgrp;
            const int gm  = m0 + M;
            const int bb  = gm >> 11;         // batch
            const int seq = gm & 2047;
            Vtws[((size_t)bb * 1024 + n0 + lhalf) * 2048 + seq] = f2bf_bits(acc[v]);
        }
    }
}

// ---------------------------------------------------------------------------
// Kernel 2: flash attention. Block = 8 waves = one 16-query tile of batch b.
// Wave w owns d-slice [w*128, w*128+128).
// Per 32-key chunk (two 16-key groups):
//   - async-copy the 16x1024 bf16 K slab into LDS (global_load_async_to_lds,
//     ASYNCcnt), overlapped with zeroing the S tile
//   - S(16x16) partials: 4 WMMAs per wave from LDS, ds_add_f32 reduce
//   - online softmax (16 threads), P in LDS
//   - O += P @ Vchunk: 8 WMMAs per wave over its 128 output columns
// ---------------------------------------------------------------------------
__global__ __launch_bounds__(256) void flash_attn_kernel(
    const unsigned short* __restrict__ Q,
    const unsigned short* __restrict__ K,
    const unsigned short* __restrict__ Vt,
    float* __restrict__ out)
{
    __shared__ __align__(16) unsigned short Ksh[16 * 1024]; // 32 KB K slab
    __shared__ float S[16][32];    // fp32 score tile (cross-wave reduced)
    __shared__ float Pf[16][32];   // exp-weights
    __shared__ float mrow[16], lrow[16], crow[16];

    const int tid   = threadIdx.x;
    const int lane  = tid & 31;
    const int wave  = tid >> 5;
    const int lhalf = lane & 15;
    const int kgrp  = lane >> 4;

    const int q0     = blockIdx.x * 16;
    const int b      = blockIdx.y;
    const int dbase0 = wave * 128;
    const float SCALE = 0.03125f;   // 1/sqrt(1024)

    // LDS byte offset of the staging slab (addrspace(3) offset = low 32 bits)
    const unsigned ksh_base = (unsigned)(size_t)(const void*)Ksh;

    // Preload this wave's 4 Q A-fragments (16 rows x 32 d each) once.
    FragBF qfrag[4];
    const unsigned short* qrow = Q + (size_t)(b * 2048 + q0 + lhalf) * 1024;
#pragma unroll
    for (int f = 0; f < 4; ++f) {
        const unsigned short* p = qrow + dbase0 + f * 32;
        qfrag[f].q[0] = *(const u32x4*)(p + kgrp * 8);
        qfrag[f].q[1] = *(const u32x4*)(p + 16 + kgrp * 8);
    }

    v8f oacc[8] = {};   // 16 queries x 128 cols, fp32

    if (tid < 16) { mrow[tid] = -__builtin_inff(); lrow[tid] = 0.0f; }
    __syncthreads();

    const int jend = (q0 + 15) >> 5;          // causal: chunks up to diagonal
    for (int j = 0; j <= jend; ++j) {
#pragma unroll
        for (int g = 0; g < 2; ++g) {         // two 16-key column groups
            // ---- async-stage K group g (16 keys x 1024 d bf16 = 32 KB) ----
            {
                const size_t gbase =
                    (size_t)(b * 2048 + j * 32 + g * 16) * 1024; // ushort idx
#pragma unroll
                for (int i = 0; i < 8; ++i) {
                    const int c = i * 256 + tid;           // 16-byte chunk id
                    const unsigned long long gp = (unsigned long long)(size_t)
                        (K + gbase + (size_t)c * 8);
                    const unsigned loff = ksh_base + (unsigned)c * 16;
                    asm volatile(
                        "global_load_async_to_lds_b128 %0, %1, off"
                        :: "v"(loff), "v"(gp) : "memory");
                }
            }
            if (g == 0) {                     // overlap with the async copy
                float* sp = &S[0][0];
                sp[tid] = 0.0f; sp[tid + 256] = 0.0f;
            }
            asm volatile("s_wait_asynccnt 0" ::: "memory");
            __syncthreads();

            // ---- partial scores over this wave's d-slice, from LDS ----
            v8f sacc = {};
            const unsigned short* krow = Ksh + lhalf * 1024 + dbase0;
#pragma unroll
            for (int f = 0; f < 4; ++f) {
                FragBF bf;
                const unsigned short* p = krow + f * 32 + kgrp * 8;
                bf.q[0] = *(const u32x4*)p;          // ds_load_b128
                bf.q[1] = *(const u32x4*)(p + 16);
                sacc = wmma_bf16(qfrag[f].v, bf.v, sacc);
            }
#pragma unroll
            for (int v = 0; v < 8; ++v)              // ds_add_f32 reduce
                atomicAdd(&S[v + 8 * kgrp][g * 16 + lhalf], sacc[v]);
            __syncthreads();   // S done for g; Ksh reads done before restage
        }

        // ---- online softmax (one thread per query row) ----
        if (tid < 16) {
            const int qg = q0 + tid;
            float m = mrow[tid], mx = m;
            for (int kk = 0; kk < 32; ++kk) {
                const int key = j * 32 + kk;
                const float s = (key <= qg) ? S[tid][kk] * SCALE
                                            : -__builtin_inff();
                mx = fmaxf(mx, s);
            }
            const float c = __expf(m - mx);
            float l = lrow[tid] * c;
            for (int kk = 0; kk < 32; ++kk) {
                const int key = j * 32 + kk;
                const float s = (key <= qg) ? S[tid][kk] * SCALE
                                            : -__builtin_inff();
                const float p = __expf(s - mx);
                Pf[tid][kk] = p;
                l += p;
            }
            mrow[tid] = mx; lrow[tid] = l; crow[tid] = c;
        }
        __syncthreads();

        // ---- rescale accumulators + O += P @ Vchunk ----
        FragBF pf;                             // P as 16x32 bf16 A-fragment
#pragma unroll
        for (int e = 0; e < 8; ++e) {
            pf.h[e]     = (__bf16)Pf[lhalf][kgrp * 8 + e];
            pf.h[e + 8] = (__bf16)Pf[lhalf][16 + kgrp * 8 + e];
        }
#pragma unroll
        for (int t = 0; t < 8; ++t) {
            const int col = dbase0 + t * 16 + lhalf;
            // B[k][n] = Vt[b][col][key]: two runs of 8 bf16 along seq, 16 apart
            const unsigned short* vp =
                Vt + ((size_t)b * 1024 + col) * 2048 + j * 32 + kgrp * 8;
            FragBF vf;
            vf.q[0] = *(const u32x4*)vp;
            vf.q[1] = *(const u32x4*)(vp + 16);
#pragma unroll
            for (int v = 0; v < 8; ++v) oacc[t][v] *= crow[v + 8 * kgrp];
            oacc[t] = wmma_bf16(pf.v, vf.v, oacc[t]);
        }
        __syncthreads();
    }

    // ---- epilogue: out = O / l ----
#pragma unroll
    for (int t = 0; t < 8; ++t) {
#pragma unroll
        for (int v = 0; v < 8; ++v) {
            const int M = v + 8 * kgrp;
            out[(size_t)(b * 2048 + q0 + M) * 1024 + dbase0 + t * 16 + lhalf] =
                oacc[t][v] / lrow[M];
        }
    }
}

// ---------------------------------------------------------------------------
extern "C" void kernel_launch(void* const* d_in, const int* in_sizes, int n_in,
                              void* d_out, int out_size, void* d_ws, size_t ws_size,
                              hipStream_t stream)
{
    (void)in_sizes; (void)n_in; (void)out_size; (void)ws_size;

    const float* x  = (const float*)d_in[0];
    const float* Wq = (const float*)d_in[1];
    const float* Wk = (const float*)d_in[2];
    const float* Wv = (const float*)d_in[3];
    float* out = (float*)d_out;

    unsigned short* Qws = (unsigned short*)d_ws;                 // 16 MB
    unsigned short* Kws = Qws + (size_t)8192 * 1024;             // 16 MB
    unsigned short* Vt  = Kws + (size_t)8192 * 1024;             // 16 MB

    // 512 m-tiles x 16 (4 n-tiles per block) x 3 matrices
    qkv_gemm_kernel<<<dim3(512, 16, 3), 128, 0, stream>>>(
        x, Wq, Wk, Wv, Qws, Kws, Vt);

    // 128 query tiles x 4 batches, 8 waves per block
    flash_attn_kernel<<<dim3(128, 4), 256, 0, stream>>>(Qws, Kws, Vt, out);
}
